// MultiHeadSelfAttention_22668837388550
// MI455X (gfx1250) — matrix operand order repaired and checked by reading, not compiled, not run on hardware
//
#include <hip/hip_runtime.h>

// ---------------------------------------------------------------------------
// MHSA for MI455X (gfx1250): bf16 WMMA everywhere, flash-attention streaming,
// TDM (tensor_load_to_lds) for K tiles, 2-stage ping-pong pipelined GEMMs.
// ---------------------------------------------------------------------------

typedef __attribute__((ext_vector_type(16))) __bf16 v16bf;
typedef __attribute__((ext_vector_type(8)))  float  v8f;
typedef __attribute__((ext_vector_type(4)))  unsigned int u32x4;
typedef __attribute__((ext_vector_type(8)))  int  i32x8;
typedef __attribute__((ext_vector_type(4)))  int  i32x4;

#define D_MODEL 768
#define N_HEADS 12
#define DK      64

#if __has_builtin(__builtin_amdgcn_tensor_load_to_lds) && __has_builtin(__builtin_amdgcn_s_wait_tensorcnt)
#define HAVE_TDM 1
#else
#define HAVE_TDM 0
#endif

__device__ __forceinline__ unsigned short f32_to_bf16(float f) {
    unsigned int u = __float_as_uint(f);
    u += 0x7FFFu + ((u >> 16) & 1u);   // round-to-nearest-even
    return (unsigned short)(u >> 16);
}

union Frag16 { uint4 q[2]; v16bf v; };

// A-matrix 16x32 bf16 fragment from row-major (K-contiguous) storage.
// lane L: row M = L%16; halves [K0..7,K16..23] (L<16) or [K8..15,K24..31].
__device__ __forceinline__ v16bf load_a_frag(const unsigned short* base, int ld, int k) {
    const int lane = threadIdx.x & 31;
    const int m    = lane & 15;
    const int koff = (lane >> 4) ? 8 : 0;
    Frag16 f;
    f.q[0] = *(const uint4*)(base + (size_t)m * ld + k + koff);
    f.q[1] = *(const uint4*)(base + (size_t)m * ld + k + 16 + koff);
    return f.v;
}

// B-matrix 32x16 bf16 fragment; column N = lane%16 maps to a K-contiguous row
// of `base` (works for W stored [N,K], K-tiles stored [key,d], Vt [d,key]).
// lane L holds halves [K0..7,K16..23] (L<16) or [K8..15,K24..31].
__device__ __forceinline__ v16bf load_b_frag(const unsigned short* base, int ld, int k) {
    const int lane = threadIdx.x & 31;
    const int n    = lane & 15;
    const int koff = (lane >> 4) ? 8 : 0;
    Frag16 f;
    f.q[0] = *(const uint4*)(base + (size_t)n * ld + k + koff);
    f.q[1] = *(const uint4*)(base + (size_t)n * ld + k + 16 + koff);
    return f.v;
}

__device__ __forceinline__ v8f wmma_bf16(v16bf a, v16bf b, v8f c) {
    return __builtin_amdgcn_wmma_f32_16x16x32_bf16(false, a, false, b, (short)0, c, false, false);
}

#if HAVE_TDM
// TDM: DMA a 64x64 bf16 tile (row stride `stride_elems`) from global to LDS.
// D# layout per CDNA5 ISA (2-D tensor == tile window).
__device__ __forceinline__ void tdm_load_tile_64x64_bf16(const unsigned short* gsrc,
                                                         unsigned lds_off,
                                                         int stride_elems) {
    const unsigned long long ga = (unsigned long long)(size_t)gsrc;
    u32x4 g0;
    g0[0] = 1u;                                   // count=1, user descriptor
    g0[1] = lds_off;                              // lds_addr (bytes)
    g0[2] = (unsigned)(ga & 0xffffffffu);         // global_addr[31:0]
    g0[3] = (unsigned)((ga >> 32) & 0x01ffffffu)  // global_addr[56:32]
            | (2u << 30);                         // type = 2 ("image")
    i32x8 g1;
    g1[0] = (int)(1u << 16);                      // data_size=1 (2 bytes/elem)
    g1[1] = (int)(64u << 16);                     // tensor_dim0 = 64
    g1[2] = (int)(64u << 16);                     // tensor_dim1 = 64
    g1[3] = (int)(64u << 16);                     // tile_dim0 = 64
    g1[4] = 64;                                   // tile_dim1 = 64, tile_dim2 = 0
    g1[5] = stride_elems;                         // tensor_dim0_stride[31:0]
    g1[6] = 0;
    g1[7] = 0;
    i32x4 z4 = {0, 0, 0, 0};
#if __clang_major__ >= 23
    i32x8 z8 = {0, 0, 0, 0, 0, 0, 0, 0};
    __builtin_amdgcn_tensor_load_to_lds(g0, g1, z4, z4, z8, 0);
#else
    __builtin_amdgcn_tensor_load_to_lds(g0, g1, z4, z4, 0);
#endif
}
#endif

// ---------------------------------------------------------------------------
// fp32 -> bf16 conversion
// ---------------------------------------------------------------------------
__global__ void cvt_f32_bf16_kernel(const float* __restrict__ in,
                                    unsigned short* __restrict__ out, int n) {
    int i = blockIdx.x * 256 + threadIdx.x;
    if (i < n) out[i] = f32_to_bf16(in[i]);
}

// ---------------------------------------------------------------------------
// C[M,N] = A[M,K] @ W[N,K]^T, bf16 in, fp32 accum.
// Block: 128 threads (4 waves). Block tile 128x64; wave w -> rows 32w..32w+31
// (two 16-row sub-blocks reusing each B fragment twice -> 8 WMMAs / fetch).
// 2-stage ping-pong: loads write directly into next stage's registers
// (no register rotation moves), WMMAs overlap in-flight loads.
// Requires K % 64 == 0 (K = 768 here).
// out_hm != nullptr : write bf16 head-major [H, S, 64]
// out_f32 != nullptr: write fp32 row-major [M, N]
// ---------------------------------------------------------------------------
__global__ __launch_bounds__(128)
void gemm_bf16_kernel(const unsigned short* __restrict__ A,
                      const unsigned short* __restrict__ W,
                      unsigned short* __restrict__ out_hm,
                      float* __restrict__ out_f32,
                      int M, int N, int K, int S) {
    const int m0   = blockIdx.y * 128;
    const int n0   = blockIdx.x * 64;
    const int wid  = threadIdx.x >> 5;
    const int lane = threadIdx.x & 31;
    const int ln   = lane & 15;
    const int hs   = lane >> 4;

    const unsigned short* ab0 = A + (size_t)(m0 + wid * 32) * K;        // rows r0..r0+15
    const unsigned short* ab1 = ab0 + (size_t)16 * K;                   // rows r0+16..r0+31
    const unsigned short* wr[4];
    #pragma unroll
    for (int s = 0; s < 4; ++s) wr[s] = W + (size_t)(n0 + s * 16) * K;

    v8f c[8];
    #pragma unroll
    for (int s = 0; s < 8; ++s) c[s] = (v8f){0.f,0.f,0.f,0.f,0.f,0.f,0.f,0.f};

    const int nk = K >> 5;                       // assumed even (K=768 -> 24)

    // stage-0 prologue
    v16bf a0[2], b0[4];
    a0[0] = load_a_frag(ab0, K, 0);
    a0[1] = load_a_frag(ab1, K, 0);
    #pragma unroll
    for (int s = 0; s < 4; ++s) b0[s] = load_b_frag(wr[s], K, 0);

    for (int ks = 0; ks < nk; ks += 2) {
        const int k1 = (ks + 1) * 32;                         // always < K (nk even)
        const int k2 = (ks + 2 < nk) ? (ks + 2) * 32 : 0;     // wrap: harmless reload
        if (ks + 4 < nk) {
            __builtin_prefetch(ab0 + (size_t)ln * K + (ks + 4) * 32, 0, 3);  // global_prefetch_b8
            __builtin_prefetch(wr[0] + (size_t)ln * K + (ks + 4) * 32, 0, 3);
        }
        // stage 1 loads (independent of stage-0 WMMAs)
        v16bf a1[2], b1[4];
        a1[0] = load_a_frag(ab0, K, k1);
        a1[1] = load_a_frag(ab1, K, k1);
        #pragma unroll
        for (int s = 0; s < 4; ++s) b1[s] = load_b_frag(wr[s], K, k1);
        // stage 0 compute
        #pragma unroll
        for (int s = 0; s < 4; ++s) {
            c[s]     = wmma_bf16(a0[0], b0[s], c[s]);
            c[4 + s] = wmma_bf16(a0[1], b0[s], c[4 + s]);
        }
        // stage 0 loads for next iteration (direct into a0/b0, no copies)
        a0[0] = load_a_frag(ab0, K, k2);
        a0[1] = load_a_frag(ab1, K, k2);
        #pragma unroll
        for (int s = 0; s < 4; ++s) b0[s] = load_b_frag(wr[s], K, k2);
        // stage 1 compute
        #pragma unroll
        for (int s = 0; s < 4; ++s) {
            c[s]     = wmma_bf16(a1[0], b1[s], c[s]);
            c[4 + s] = wmma_bf16(a1[1], b1[s], c[4 + s]);
        }
    }

    #pragma unroll
    for (int sb = 0; sb < 2; ++sb) {
        #pragma unroll
        for (int s = 0; s < 4; ++s) {
            #pragma unroll
            for (int v = 0; v < 8; ++v) {
                const int m = m0 + wid * 32 + sb * 16 + v + 8 * hs;
                const int n = n0 + s * 16 + ln;
                const float val = c[sb * 4 + s][v];
                if (out_f32) {
                    out_f32[(size_t)m * N + n] = val;
                } else {
                    const int h = n >> 6, d = n & 63;
                    out_hm[((size_t)h * S + m) * DK + d] = f32_to_bf16(val);
                }
            }
        }
    }
}

// ---------------------------------------------------------------------------
// Causal flash attention. Grid: (S/64, H). Block: 128 threads (4 waves).
// Wave w owns query rows q0+16w .. q0+16w+15. Streams 64-key blocks.
// K tile DMA'd via TDM (wave 0, TENSORcnt); V transposed cooperatively.
// Q/K/V bf16 head-major [H,S,64]; output bf16 [S, D_MODEL] (head-interleaved).
// ---------------------------------------------------------------------------
__global__ __launch_bounds__(128)
void attn_kernel(const unsigned short* __restrict__ Qb,
                 const unsigned short* __restrict__ Kb,
                 const unsigned short* __restrict__ Vb,
                 unsigned short* __restrict__ attn_out, int S) {
    __shared__ __align__(32) unsigned short lds_k[64 * 64];      // [key][d] (first shared var: LDS offset 0)
    __shared__ __align__(32) unsigned short lds_vt[64 * 80];     // [d][key], padded
    __shared__ __align__(32) unsigned short lds_p[4][16 * 80];   // per-wave P tile

    const int h    = blockIdx.y;
    const int q0   = blockIdx.x * 64;
    const int tid  = threadIdx.x;
    const int wid  = tid >> 5;
    const int lane = tid & 31;
    const int ln   = lane & 15;
    const int hs   = lane >> 4;

    const unsigned short* Qh = Qb + (size_t)h * S * DK;
    const unsigned short* Kh = Kb + (size_t)h * S * DK;
    const unsigned short* Vh = Vb + (size_t)h * S * DK;

    // Q fragments for this wave's 16 rows (K = d, 2 chunks of 32)
    const unsigned short* qbase = Qh + (size_t)(q0 + wid * 16) * DK;
    v16bf qa0 = load_a_frag(qbase, DK, 0);
    v16bf qa1 = load_a_frag(qbase, DK, 32);

    v8f o[4];
    #pragma unroll
    for (int s = 0; s < 4; ++s) o[s] = (v8f){0.f,0.f,0.f,0.f,0.f,0.f,0.f,0.f};
    float m_st[8], l_st[8];
    #pragma unroll
    for (int v = 0; v < 8; ++v) { m_st[v] = -1e30f; l_st[v] = 0.f; }

    const float scale = 0.125f;   // 1/sqrt(64)

    for (int j0 = 0; j0 <= q0; j0 += 64) {
        __syncthreads();          // previous tile fully consumed
#if HAVE_TDM
        if (wid == 0)             // one TDM descriptor per key block (per-wave op)
            tdm_load_tile_64x64_bf16(Kh + (size_t)j0 * DK, 0u /* lds_k at LDS offset 0 */, DK);
#endif
        {   // cooperative: V tile transposed into LDS
            const int key = tid >> 1;
            const int dp  = (tid & 1) * 32;
#if !HAVE_TDM
            const uint4* gk = (const uint4*)(Kh + (size_t)(j0 + key) * DK + dp);
            uint4* sk = (uint4*)(lds_k + key * 64 + dp);
            sk[0] = gk[0]; sk[1] = gk[1]; sk[2] = gk[2]; sk[3] = gk[3];
#endif
            const unsigned short* gv = Vh + (size_t)(j0 + key) * DK + dp;
            #pragma unroll
            for (int i = 0; i < 32; ++i) lds_vt[(dp + i) * 80 + key] = gv[i];
        }
#if HAVE_TDM
        if (wid == 0) __builtin_amdgcn_s_wait_tensorcnt(0);   // TDM drained before barrier
#endif
        __syncthreads();

        // ---- S = (Q K^T) * scale (+ causal mask on diagonal block) ----
        v8f sacc[4];
        #pragma unroll
        for (int s = 0; s < 4; ++s) {
            sacc[s] = (v8f){0.f,0.f,0.f,0.f,0.f,0.f,0.f,0.f};
            v16bf bk0 = load_b_frag(lds_k + (s * 16) * 64, 64, 0);
            v16bf bk1 = load_b_frag(lds_k + (s * 16) * 64, 64, 32);
            sacc[s] = wmma_bf16(qa0, bk0, sacc[s]);
            sacc[s] = wmma_bf16(qa1, bk1, sacc[s]);
        }
        const bool diag = (j0 == q0);
        #pragma unroll
        for (int s = 0; s < 4; ++s)
            #pragma unroll
            for (int v = 0; v < 8; ++v) {
                float sv = sacc[s][v] * scale;
                if (diag) {
                    const int kcol = j0 + s * 16 + ln;
                    const int qrow = q0 + wid * 16 + v + 8 * hs;
                    if (kcol > qrow) sv += -1e9f;
                }
                sacc[s][v] = sv;
            }

        // ---- online softmax ----
        float mcur[8];
        #pragma unroll
        for (int v = 0; v < 8; ++v) {
            float mv = sacc[0][v];
            mv = fmaxf(mv, sacc[1][v]); mv = fmaxf(mv, sacc[2][v]); mv = fmaxf(mv, sacc[3][v]);
            mv = fmaxf(mv, __shfl_xor(mv, 1));
            mv = fmaxf(mv, __shfl_xor(mv, 2));
            mv = fmaxf(mv, __shfl_xor(mv, 4));
            mv = fmaxf(mv, __shfl_xor(mv, 8));
            mcur[v] = mv;
        }
        float alpha[8], rsum[8];
        #pragma unroll
        for (int v = 0; v < 8; ++v) {
            const float mnew = fmaxf(m_st[v], mcur[v]);
            alpha[v] = __expf(m_st[v] - mnew);
            m_st[v]  = mnew;
            rsum[v]  = 0.f;
        }
        unsigned short* prow = lds_p[wid];
        #pragma unroll
        for (int s = 0; s < 4; ++s)
            #pragma unroll
            for (int v = 0; v < 8; ++v) {
                const float p = __expf(sacc[s][v] - m_st[v]);
                rsum[v] += p;
                prow[(v + 8 * hs) * 80 + s * 16 + ln] = f32_to_bf16(p);
            }
        #pragma unroll
        for (int v = 0; v < 8; ++v) {
            float rv = rsum[v];
            rv += __shfl_xor(rv, 1);
            rv += __shfl_xor(rv, 2);
            rv += __shfl_xor(rv, 4);
            rv += __shfl_xor(rv, 8);
            l_st[v] = alpha[v] * l_st[v] + rv;
        }
        #pragma unroll
        for (int s = 0; s < 4; ++s)
            #pragma unroll
            for (int v = 0; v < 8; ++v) o[s][v] *= alpha[v];

        // ---- O += P @ V (P re-fragmented from per-wave LDS; DS ops in-order) ----
        v16bf pa0 = load_a_frag(prow, 80, 0);
        v16bf pa1 = load_a_frag(prow, 80, 32);
        #pragma unroll
        for (int s = 0; s < 4; ++s) {
            v16bf bv0 = load_b_frag(lds_vt + (s * 16) * 80, 80, 0);
            v16bf bv1 = load_b_frag(lds_vt + (s * 16) * 80, 80, 32);
            o[s] = wmma_bf16(pa0, bv0, o[s]);
            o[s] = wmma_bf16(pa1, bv1, o[s]);
        }
    }

    // ---- normalize and write bf16 [S, D_MODEL] (column h*64 + d) ----
    float inv_l[8];
    #pragma unroll
    for (int v = 0; v < 8; ++v) inv_l[v] = 1.0f / l_st[v];
    #pragma unroll
    for (int s = 0; s < 4; ++s)
        #pragma unroll
        for (int v = 0; v < 8; ++v) {
            const int row = q0 + wid * 16 + v + 8 * hs;
            const int col = h * DK + s * 16 + ln;
            attn_out[(size_t)row * D_MODEL + col] = f32_to_bf16(o[s][v] * inv_l[v]);
        }
}

// ---------------------------------------------------------------------------
// Host side
// ---------------------------------------------------------------------------
extern "C" void kernel_launch(void* const* d_in, const int* in_sizes, int n_in,
                              void* d_out, int out_size, void* d_ws, size_t ws_size,
                              hipStream_t stream) {
    const float* x  = (const float*)d_in[0];
    const float* Wq = (const float*)d_in[1];
    const float* Wk = (const float*)d_in[2];
    const float* Wv = (const float*)d_in[3];
    const float* Wo = (const float*)d_in[4];
    float* out = (float*)d_out;

    const int S  = in_sizes[0] / D_MODEL;      // 4096
    const int SD = S * D_MODEL;
    const int DD = D_MODEL * D_MODEL;

    unsigned short* xb   = (unsigned short*)d_ws;
    unsigned short* wqb  = xb  + SD;
    unsigned short* wkb  = wqb + DD;
    unsigned short* wvb  = wkb + DD;
    unsigned short* wob  = wvb + DD;
    unsigned short* Qh   = wob + DD;           // head-major [H,S,64]
    unsigned short* Kh   = Qh  + SD;
    unsigned short* Vh   = Kh  + SD;
    unsigned short* attn = Vh  + SD;           // [S, D_MODEL] bf16

    // 1) fp32 -> bf16 conversions
    cvt_f32_bf16_kernel<<<(SD + 255) / 256, 256, 0, stream>>>(x,  xb,  SD);
    cvt_f32_bf16_kernel<<<(DD + 255) / 256, 256, 0, stream>>>(Wq, wqb, DD);
    cvt_f32_bf16_kernel<<<(DD + 255) / 256, 256, 0, stream>>>(Wk, wkb, DD);
    cvt_f32_bf16_kernel<<<(DD + 255) / 256, 256, 0, stream>>>(Wv, wvb, DD);
    cvt_f32_bf16_kernel<<<(DD + 255) / 256, 256, 0, stream>>>(Wo, wob, DD);

    // 2) QKV projections (bf16 head-major outputs)
    dim3 gproj(D_MODEL / 64, S / 128);
    gemm_bf16_kernel<<<gproj, 128, 0, stream>>>(xb, wqb, Qh, nullptr, S, D_MODEL, D_MODEL, S);
    gemm_bf16_kernel<<<gproj, 128, 0, stream>>>(xb, wkb, Kh, nullptr, S, D_MODEL, D_MODEL, S);
    gemm_bf16_kernel<<<gproj, 128, 0, stream>>>(xb, wvb, Vh, nullptr, S, D_MODEL, D_MODEL, S);

    // 3) causal flash attention
    dim3 gattn(S / 64, N_HEADS);
    attn_kernel<<<gattn, 128, 0, stream>>>(Qh, Kh, Vh, attn, S);

    // 4) output projection (fp32 out)
    gemm_bf16_kernel<<<gproj, 128, 0, stream>>>(attn, wob, nullptr, out, S, D_MODEL, D_MODEL, S);
}
